// SparseMLP_2018634629287
// MI455X (gfx1250) — compile-verified, matrix operand order checked by reading, not run on hardware
//
#include <hip/hip_runtime.h>
#include <hip/hip_bf16.h>

typedef __attribute__((ext_vector_type(16))) __bf16 v16bf;
typedef __attribute__((ext_vector_type(8)))  __bf16 v8bf;
typedef __attribute__((ext_vector_type(4)))  __bf16 v4bf;
typedef __attribute__((ext_vector_type(8)))  float  v8f;
typedef __attribute__((ext_vector_type(4)))  float  v4f;

#define BM 128
#define BN 128
#define BK 64
#define LDK (BK + 8)   // 72 bf16 = 144B rows: b128 frag reads are bank-conflict-free

#define T_TOT 8192
#define HID   2048
#define FFN   8192
#define NEXP  8
#define TPE   (T_TOT / NEXP)   // 1024 tokens per expert

__device__ __forceinline__ v16bf cat8(v8bf lo, v8bf hi) {
  return __builtin_shufflevector(lo, hi, 0,1,2,3,4,5,6,7,8,9,10,11,12,13,14,15);
}

// Branch-free tanh-GELU: 0.5x(1+tanh(t)) == x * sigmoid(2t),
// sigmoid(2t) = 1/(1+exp(-2t)) = 1/(1 + 2^(-2t*log2 e)).
// Uses v_exp_f32 / v_rcp_f32 (TRANS pipe, co-executes with WMMA).
__device__ __forceinline__ float gelu_tanh(float x) {
  float t = x + 0.044715f * x * x * x;
  float z = -2.302208131700935f * t;     // -2*0.7978845608*log2(e) * t
  float e = __builtin_amdgcn_exp2f(z);
  return x * __builtin_amdgcn_rcpf(1.0f + e);
}

// Stage B tile (f32 [BK][BN] row-major in global) into LDS transposed [n][k]
// as bf16, via per-thread 4x4 micro-tile register transpose + ds_store_b64.
template <int LDB>
__device__ __forceinline__ void stage_B(const float* __restrict__ Bg,
                                        __bf16 (*Bs)[LDK], int tid) {
#pragma unroll
  for (int it = 0; it < 2; ++it) {
    int idx = tid + it * 256;            // 0..511
    int k   = (idx >> 5) << 2;           // 0,4,...,60
    int n   = (idx & 31) << 2;           // 0,4,...,124
    const float* p = Bg + (size_t)k * LDB + n;
    v4f r0 = *(const v4f*)(p);
    v4f r1 = *(const v4f*)(p + LDB);
    v4f r2 = *(const v4f*)(p + 2 * LDB);
    v4f r3 = *(const v4f*)(p + 3 * LDB);
    v4bf c0 = {(__bf16)r0.x, (__bf16)r1.x, (__bf16)r2.x, (__bf16)r3.x};
    v4bf c1 = {(__bf16)r0.y, (__bf16)r1.y, (__bf16)r2.y, (__bf16)r3.y};
    v4bf c2 = {(__bf16)r0.z, (__bf16)r1.z, (__bf16)r2.z, (__bf16)r3.z};
    v4bf c3 = {(__bf16)r0.w, (__bf16)r1.w, (__bf16)r2.w, (__bf16)r3.w};
    *(v4bf*)&Bs[n + 0][k] = c0;
    *(v4bf*)&Bs[n + 1][k] = c1;
    *(v4bf*)&Bs[n + 2][k] = c2;
    *(v4bf*)&Bs[n + 3][k] = c3;
  }
}

// ---------------------------------------------------------------------------
// GEMM1 + GELU:  h[e] = gelu( X[e] (1024x2048 f32) @ W1[e] (2048x8192 f32) )
// stored bf16 to workspace. grid = (FFN/BN, TPE/BM, E), block = 256 (8 waves).
// ---------------------------------------------------------------------------
__global__ __launch_bounds__(256)
void moe_gemm1_gelu(const float* __restrict__ X, const float* __restrict__ W1,
                    __hip_bfloat16* __restrict__ Hws) {
  const int e  = blockIdx.z;
  const int m0 = blockIdx.y * BM;
  const int n0 = blockIdx.x * BN;

  const float* Xe  = X  + (size_t)e * TPE * HID;
  const float* W1e = W1 + (size_t)e * HID * FFN;
  __hip_bfloat16* He = Hws + (size_t)e * TPE * FFN;

  __shared__ __bf16 As[BM][LDK];   // [row][k]
  __shared__ __bf16 Bs[BN][LDK];   // [col][k] (transposed at staging)

  const int tid   = threadIdx.x;
  const int lane  = tid & 31;
  const int wave  = tid >> 5;
  const int wr    = wave >> 1;       // 0..3  -> 32-row band
  const int wc    = wave & 1;        // 0..1  -> 64-col band
  const int half  = lane >> 4;
  const int l16   = lane & 15;
  const int kbase = half * 8;        // per documented 16-bit A/B lane layout

  v8f acc[2][4];
#pragma unroll
  for (int i = 0; i < 2; ++i)
#pragma unroll
    for (int j = 0; j < 4; ++j) acc[i][j] = v8f{};

  for (int k0 = 0; k0 < HID; k0 += BK) {
    // Stage A: 128x64 f32 -> bf16, packed ds_store_b64. 8 float4 per thread.
#pragma unroll
    for (int it = 0; it < 8; ++it) {
      int idx = tid + it * 256;          // 0..2047
      int r   = idx >> 4;                // 0..127
      int kv  = (idx & 15) << 2;         // 0,4,...,60
      v4f a4 = *(const v4f*)(Xe + (size_t)(m0 + r) * HID + k0 + kv);
      v4bf p = {(__bf16)a4.x, (__bf16)a4.y, (__bf16)a4.z, (__bf16)a4.w};
      *(v4bf*)&As[r][kv] = p;
    }
    stage_B<FFN>(W1e + (size_t)k0 * FFN + n0, Bs, tid);
    __syncthreads();

#pragma unroll
    for (int s = 0; s < 2; ++s) {        // two K=32 WMMA steps per LDS tile
      const int kb = s * 32 + kbase;
#pragma unroll
      for (int i = 0; i < 2; ++i) {
        int ar = wr * 32 + i * 16 + l16;
        v16bf a = cat8(*(const v8bf*)&As[ar][kb],
                       *(const v8bf*)&As[ar][kb + 16]);
#pragma unroll
        for (int j = 0; j < 4; ++j) {
          int bc = wc * 64 + j * 16 + l16;
          v16bf b = cat8(*(const v8bf*)&Bs[bc][kb],
                         *(const v8bf*)&Bs[bc][kb + 16]);
          acc[i][j] = __builtin_amdgcn_wmma_f32_16x16x32_bf16(
              false, a, false, b, (short)0, acc[i][j], false, false);
        }
      }
    }
    __syncthreads();
  }

  // Epilogue: branch-free GELU + bf16 store.
#pragma unroll
  for (int i = 0; i < 2; ++i) {
#pragma unroll
    for (int j = 0; j < 4; ++j) {
      int col = n0 + wc * 64 + j * 16 + l16;
#pragma unroll
      for (int v = 0; v < 8; ++v) {
        int row = m0 + wr * 32 + i * 16 + half * 8 + v;
        He[(size_t)row * FFN + col] = __float2bfloat16(gelu_tanh(acc[i][j][v]));
      }
    }
  }
}

// ---------------------------------------------------------------------------
// GEMM2:  out[e] = h[e] (1024x8192 bf16) @ W2[e] (8192x2048 f32)
// grid = (HID/BN, TPE/BM, E), block = 256.
// ---------------------------------------------------------------------------
__global__ __launch_bounds__(256)
void moe_gemm2(const __hip_bfloat16* __restrict__ Hws,
               const float* __restrict__ W2, float* __restrict__ Out) {
  const int e  = blockIdx.z;
  const int m0 = blockIdx.y * BM;
  const int n0 = blockIdx.x * BN;

  const __bf16* He = (const __bf16*)Hws + (size_t)e * TPE * FFN;
  const float* W2e = W2 + (size_t)e * FFN * HID;
  float* Oe = Out + (size_t)e * TPE * HID;

  __shared__ __bf16 As[BM][LDK];
  __shared__ __bf16 Bs[BN][LDK];

  const int tid   = threadIdx.x;
  const int lane  = tid & 31;
  const int wave  = tid >> 5;
  const int wr    = wave >> 1;
  const int wc    = wave & 1;
  const int half  = lane >> 4;
  const int l16   = lane & 15;
  const int kbase = half * 8;

  v8f acc[2][4];
#pragma unroll
  for (int i = 0; i < 2; ++i)
#pragma unroll
    for (int j = 0; j < 4; ++j) acc[i][j] = v8f{};

  for (int k0 = 0; k0 < FFN; k0 += BK) {
    // Stage A: 128x64 bf16 (already bf16 in ws): 16B copies, 4 per thread.
#pragma unroll
    for (int it = 0; it < 4; ++it) {
      int idx = tid + it * 256;          // 0..1023
      int r   = idx >> 3;                // 0..127
      int c   = (idx & 7) << 3;          // 0,8,...,56
      v8bf a8 = *(const v8bf*)(He + (size_t)(m0 + r) * FFN + k0 + c);
      *(v8bf*)&As[r][c] = a8;
    }
    stage_B<HID>(W2e + (size_t)k0 * HID + n0, Bs, tid);
    __syncthreads();

#pragma unroll
    for (int s = 0; s < 2; ++s) {
      const int kb = s * 32 + kbase;
#pragma unroll
      for (int i = 0; i < 2; ++i) {
        int ar = wr * 32 + i * 16 + l16;
        v16bf a = cat8(*(const v8bf*)&As[ar][kb],
                       *(const v8bf*)&As[ar][kb + 16]);
#pragma unroll
        for (int j = 0; j < 4; ++j) {
          int bc = wc * 64 + j * 16 + l16;
          v16bf b = cat8(*(const v8bf*)&Bs[bc][kb],
                         *(const v8bf*)&Bs[bc][kb + 16]);
          acc[i][j] = __builtin_amdgcn_wmma_f32_16x16x32_bf16(
              false, a, false, b, (short)0, acc[i][j], false, false);
        }
      }
    }
    __syncthreads();
  }

#pragma unroll
  for (int i = 0; i < 2; ++i) {
#pragma unroll
    for (int j = 0; j < 4; ++j) {
      int col = n0 + wc * 64 + j * 16 + l16;
#pragma unroll
      for (int v = 0; v < 8; ++v) {
        int row = m0 + wr * 32 + i * 16 + half * 8 + v;
        Oe[(size_t)row * HID + col] = acc[i][j][v];
      }
    }
  }
}

extern "C" void kernel_launch(void* const* d_in, const int* in_sizes, int n_in,
                              void* d_out, int out_size, void* d_ws, size_t ws_size,
                              hipStream_t stream) {
  const float* x  = (const float*)d_in[0];   // [8192, 2048] f32
  const float* w1 = (const float*)d_in[1];   // [8, 2048, 8192] f32
  const float* w2 = (const float*)d_in[2];   // [8, 8192, 2048] f32
  __hip_bfloat16* h = (__hip_bfloat16*)d_ws; // [8, 1024, 8192] bf16 = 128 MiB
  float* out = (float*)d_out;                // [8192, 2048] f32

  dim3 blk(256, 1, 1);
  dim3 g1(FFN / BN, TPE / BM, NEXP);  // (64, 8, 8)
  dim3 g2(HID / BN, TPE / BM, NEXP);  // (16, 8, 8)
  moe_gemm1_gelu<<<g1, blk, 0, stream>>>(x, w1, h);
  moe_gemm2<<<g2, blk, 0, stream>>>(h, w2, out);
}
